// MultiHeadAttention_28269474742987
// MI455X (gfx1250) — compile-verified, matrix operand order checked by reading, not run on hardware
//
#include <hip/hip_runtime.h>
#include <hip/hip_bf16.h>

#define B_  2
#define T_  2048
#define E_  1024
#define H_  16
#define DH_ 64

typedef __attribute__((ext_vector_type(16))) _Float16 v16h;
typedef __attribute__((ext_vector_type(8)))  _Float16 v8h;
typedef __attribute__((ext_vector_type(8)))  float    v8f;

#define WMMA_F16(A, Bm, C) \
  __builtin_amdgcn_wmma_f32_16x16x32_f16(false, (A), false, (Bm), (short)0, (C), false, false)

// ---------------------------------------------------------------------------
// Fragment load: 16x32 f16 tile, A-layout (also used for B^T tiles since the
// B operand mirrors A on the transposed matrix).
//   lanes 0-15 : row = lane,    elems 0-7 = K 0..7,   elems 8-15 = K 16..23
//   lanes16-31 : row = lane-16, elems 0-7 = K 8..15,  elems 8-15 = K 24..31
// Two contiguous 16-byte loads per lane.
// ---------------------------------------------------------------------------
static __device__ __forceinline__ v16h frag_ld(const _Float16* __restrict__ base, int ld) {
  const int lane = threadIdx.x & 31;
  const int row  = lane & 15;
  const int kofs = (lane >> 4) << 3;           // 0 or 8
  const _Float16* p = base + (size_t)row * ld + kofs;
  v8h lo = *reinterpret_cast<const v8h*>(p);
  v8h hi = *reinterpret_cast<const v8h*>(p + 16);
  v16h r;
#pragma unroll
  for (int i = 0; i < 8; ++i) { r[i] = lo[i]; r[8 + i] = hi[i]; }
  return r;
}

// ---------------------------------------------------------------------------
// Elementwise f32 -> f16 convert
// ---------------------------------------------------------------------------
__global__ void cvt_f32_f16(const float* __restrict__ in, _Float16* __restrict__ out, int n) {
  int i = blockIdx.x * blockDim.x + threadIdx.x;
  if (i < n) out[i] = (_Float16)in[i];
}

// [H,E,Dh] f32 -> [H,Dh,E] f16 (transpose inner two dims)
__global__ void transpose_w(const float* __restrict__ in, _Float16* __restrict__ out) {
  int i = blockIdx.x * blockDim.x + threadIdx.x;   // over H*Dh*E
  int e = i % E_;
  int d = (i / E_) % DH_;
  int h = i / (E_ * DH_);
  out[i] = (_Float16)in[((size_t)h * E_ + e) * DH_ + d];
}

// ---------------------------------------------------------------------------
// Fused QKV projection, software-pipelined k-loop. One wave -> one 16(t)x16(d)
// tile of Q,K,V for one head; A fragment (x) shared by all three WMMAs.
// Q scaled by Dh^-0.5. V stored TRANSPOSED [B,H,Dh,T] for the PV B-fragment.
// grid (T/16, H, B), block 128 (4 waves = 4 d-tiles, Dh=64).
// ---------------------------------------------------------------------------
__global__ __launch_bounds__(128) void qkv_gemm(
    const _Float16* __restrict__ xh,
    const _Float16* __restrict__ wqt, const _Float16* __restrict__ wkt,
    const _Float16* __restrict__ wvt,
    _Float16* __restrict__ Q, _Float16* __restrict__ K, _Float16* __restrict__ Vt) {
  const int wv   = threadIdx.x >> 5;     // d-tile 0..3
  const int lane = threadIdx.x & 31;
  const int tt = blockIdx.x, h = blockIdx.y, b = blockIdx.z;

  const _Float16* xa = xh  + ((size_t)b * T_ + tt * 16) * E_;
  const _Float16* bq = wqt + ((size_t)h * DH_ + wv * 16) * E_;
  const _Float16* bk = wkt + ((size_t)h * DH_ + wv * 16) * E_;
  const _Float16* bv = wvt + ((size_t)h * DH_ + wv * 16) * E_;

  v8f aq = {}, ak = {}, av = {};

  // software pipeline: stage i+1 loads issued before stage i WMMAs
  v16h a  = frag_ld(xa, E_);
  v16h fq = frag_ld(bq, E_);
  v16h fk = frag_ld(bk, E_);
  v16h fv = frag_ld(bv, E_);
  for (int e0 = 32; e0 < E_; e0 += 32) {
    v16h a_n  = frag_ld(xa + e0, E_);
    v16h fq_n = frag_ld(bq + e0, E_);
    v16h fk_n = frag_ld(bk + e0, E_);
    v16h fv_n = frag_ld(bv + e0, E_);
    aq = WMMA_F16(a, fq, aq);
    ak = WMMA_F16(a, fk, ak);
    av = WMMA_F16(a, fv, av);
    a = a_n; fq = fq_n; fk = fk_n; fv = fv_n;
  }
  aq = WMMA_F16(a, fq, aq);
  ak = WMMA_F16(a, fk, ak);
  av = WMMA_F16(a, fv, av);

  const int n = lane & 15;
  const int mb = (lane >> 4) * 8;                // D-layout: row = r + mb
  const size_t headoff = ((size_t)b * H_ + h) * (size_t)T_ * DH_;
#pragma unroll
  for (int r = 0; r < 8; ++r) {
    int t = tt * 16 + mb + r;
    Q[headoff + (size_t)t * DH_ + wv * 16 + n] = (_Float16)(aq[r] * 0.125f); // Dh^-0.5
    K[headoff + (size_t)t * DH_ + wv * 16 + n] = (_Float16)ak[r];
  }
  // V transposed: column d = wv*16+n fixed per lane, rows t consecutive -> one b128 store
  v8h vp;
#pragma unroll
  for (int r = 0; r < 8; ++r) vp[r] = (_Float16)av[r];
  _Float16* vd = Vt + ((size_t)b * H_ + h) * (size_t)DH_ * T_
                    + (size_t)(wv * 16 + n) * T_ + tt * 16 + mb;
  *reinterpret_cast<v8h*>(vd) = vp;
}

// ---------------------------------------------------------------------------
// Flash attention, causal. One wave owns a 16-row tile, streams 32 keys/step.
// All K/V fragment loads for a chunk are issued before the softmax VALU work
// so their latency hides under v_exp / shuffles.
// grid (T/16/4, B*H), block 128 (4 waves = 4 consecutive row tiles).
// ---------------------------------------------------------------------------
__global__ __launch_bounds__(128) void flash_attn(
    const _Float16* __restrict__ Q, const _Float16* __restrict__ K,
    const _Float16* __restrict__ Vt, _Float16* __restrict__ Oh) {
  __shared__ _Float16 pl[4][16 * 40];            // per-wave P staging, padded rows

  const int wv   = threadIdx.x >> 5;
  const int lane = threadIdx.x & 31;
  const int n    = lane & 15;
  const int mb   = (lane >> 4) * 8;

  const int tt = blockIdx.x * 4 + wv;
  const int bh = blockIdx.y;
  const int b  = bh / H_, h = bh % H_;
  const int t0 = tt * 16;

  const _Float16* Qb = Q  + (size_t)bh * T_ * DH_;
  const _Float16* Kb = K  + (size_t)bh * T_ * DH_;
  const _Float16* Vb = Vt + (size_t)bh * DH_ * T_;

  v16h qa0 = frag_ld(Qb + (size_t)t0 * DH_, DH_);
  v16h qa1 = frag_ld(Qb + (size_t)t0 * DH_ + 32, DH_);

  float mi[8], li[8];
#pragma unroll
  for (int r = 0; r < 8; ++r) { mi[r] = -3.0e38f; li[r] = 0.0f; }
  v8f acc[4] = {};

  for (int s0 = 0; s0 <= t0; s0 += 32) {         // 32 keys per step (2 S-subtiles)
    // issue all chunk loads up front
    v16h kb0 = frag_ld(Kb + (size_t)(s0)      * DH_, DH_);
    v16h kb1 = frag_ld(Kb + (size_t)(s0)      * DH_ + 32, DH_);
    v16h kb2 = frag_ld(Kb + (size_t)(s0 + 16) * DH_, DH_);
    v16h kb3 = frag_ld(Kb + (size_t)(s0 + 16) * DH_ + 32, DH_);
    v16h vb[4];
#pragma unroll
    for (int dt = 0; dt < 4; ++dt)
      vb[dt] = frag_ld(Vb + (size_t)(dt * 16) * T_ + s0, T_);   // B^T tile of V

    v8f s[2];
    {
      v8f z = {};
      z = WMMA_F16(qa0, kb0, z);
      z = WMMA_F16(qa1, kb1, z);
      s[0] = z;
      v8f y = {};
      y = WMMA_F16(qa0, kb2, y);
      y = WMMA_F16(qa1, kb3, y);
      s[1] = y;
    }
    // causal mask (per-lane select, no EXEC divergence)
#pragma unroll
    for (int j = 0; j < 2; ++j) {
      const int col = s0 + j * 16 + n;
#pragma unroll
      for (int r = 0; r < 8; ++r)
        if (col > t0 + mb + r) s[j][r] = -3.0e38f;
    }

    float alpha[8];
#pragma unroll
    for (int r = 0; r < 8; ++r) {
      float mx = fmaxf(s[0][r], s[1][r]);        // row-max across 16 lanes
      mx = fmaxf(mx, __shfl_xor(mx, 1));
      mx = fmaxf(mx, __shfl_xor(mx, 2));
      mx = fmaxf(mx, __shfl_xor(mx, 4));
      mx = fmaxf(mx, __shfl_xor(mx, 8));
      float mnew = fmaxf(mi[r], mx);
      float a  = __expf(mi[r] - mnew);
      float p0 = __expf(s[0][r] - mnew);
      float p1 = __expf(s[1][r] - mnew);
      float rs = p0 + p1;
      rs += __shfl_xor(rs, 1);
      rs += __shfl_xor(rs, 2);
      rs += __shfl_xor(rs, 4);
      rs += __shfl_xor(rs, 8);
      li[r] = li[r] * a + rs;
      mi[r] = mnew;
      alpha[r] = a;
      // stage P (D-layout -> LDS row-major); DS ops are in-order per wave
      pl[wv][(mb + r) * 40 + n]      = (_Float16)p0;
      pl[wv][(mb + r) * 40 + 16 + n] = (_Float16)p1;
    }
#pragma unroll
    for (int dt = 0; dt < 4; ++dt)
#pragma unroll
      for (int r = 0; r < 8; ++r) acc[dt][r] *= alpha[r];

    // reload P in A-fragment layout
    const _Float16* pp = &pl[wv][0] + n * 40 + ((lane >> 4) << 3);
    v8h lo = *reinterpret_cast<const v8h*>(pp);
    v8h hi = *reinterpret_cast<const v8h*>(pp + 16);
    v16h pa;
#pragma unroll
    for (int i = 0; i < 8; ++i) { pa[i] = lo[i]; pa[8 + i] = hi[i]; }

#pragma unroll
    for (int dt = 0; dt < 4; ++dt)
      acc[dt] = WMMA_F16(pa, vb[dt], acc[dt]);
  }

  // normalize and store head-concat f16 [B,T,E]
  float inv[8];
#pragma unroll
  for (int r = 0; r < 8; ++r) inv[r] = 1.0f / li[r];
#pragma unroll
  for (int dt = 0; dt < 4; ++dt)
#pragma unroll
    for (int r = 0; r < 8; ++r) {
      size_t idx = ((size_t)b * T_ + t0 + mb + r) * E_ + h * DH_ + dt * 16 + n;
      Oh[idx] = (_Float16)(acc[dt][r] * inv[r]);
    }
}

// ---------------------------------------------------------------------------
// Output projection: out = Oh @ wo_w^T + b. wo_w is [out,in] = N x K row-major,
// i.e. already the B^T layout frag_ld expects.
// Each wave computes a 16x64 strip: one A fragment feeds 4 WMMAs.
// Software-pipelined k-loop. grid ((B*T)/16, E/256), block 128.
// ---------------------------------------------------------------------------
__global__ __launch_bounds__(128) void out_proj(
    const _Float16* __restrict__ Oh, const _Float16* __restrict__ woh,
    const float* __restrict__ wob, float* __restrict__ out) {
  const int wv   = threadIdx.x >> 5;
  const int lane = threadIdx.x & 31;
  const int rt = blockIdx.x;
  const int n0 = blockIdx.y * 256 + wv * 64;     // first output column of strip

  const _Float16* A = Oh + (size_t)rt * 16 * E_;
  const _Float16* Bp[4];
#pragma unroll
  for (int dt = 0; dt < 4; ++dt) Bp[dt] = woh + (size_t)(n0 + dt * 16) * E_;

  v8f acc[4] = {};
  v16h a = frag_ld(A, E_);
  v16h w[4];
#pragma unroll
  for (int dt = 0; dt < 4; ++dt) w[dt] = frag_ld(Bp[dt], E_);

  for (int k0 = 32; k0 < E_; k0 += 32) {
    v16h a_n = frag_ld(A + k0, E_);
    v16h w_n[4];
#pragma unroll
    for (int dt = 0; dt < 4; ++dt) w_n[dt] = frag_ld(Bp[dt] + k0, E_);
#pragma unroll
    for (int dt = 0; dt < 4; ++dt) acc[dt] = WMMA_F16(a, w[dt], acc[dt]);
    a = a_n;
#pragma unroll
    for (int dt = 0; dt < 4; ++dt) w[dt] = w_n[dt];
  }
#pragma unroll
  for (int dt = 0; dt < 4; ++dt) acc[dt] = WMMA_F16(a, w[dt], acc[dt]);

  const int n = lane & 15;
  const int mb = (lane >> 4) * 8;
#pragma unroll
  for (int dt = 0; dt < 4; ++dt) {
    const float bias = wob[n0 + dt * 16 + n];
#pragma unroll
    for (int r = 0; r < 8; ++r)
      out[(size_t)(rt * 16 + mb + r) * E_ + n0 + dt * 16 + n] = acc[dt][r] + bias;
  }
}

// ---------------------------------------------------------------------------
extern "C" void kernel_launch(void* const* d_in, const int* in_sizes, int n_in,
                              void* d_out, int out_size, void* d_ws, size_t ws_size,
                              hipStream_t stream) {
  const float* x    = (const float*)d_in[0];
  const float* Wq   = (const float*)d_in[1];
  const float* Wk   = (const float*)d_in[2];
  const float* Wv   = (const float*)d_in[3];
  const float* wo_w = (const float*)d_in[4];
  const float* wo_b = (const float*)d_in[5];
  float* out = (float*)d_out;

  // workspace carve-up (bytes)
  char* ws = (char*)d_ws;
  const size_t sz_xh  = (size_t)B_ * T_ * E_ * 2;       // 8 MB
  const size_t sz_wt  = (size_t)H_ * DH_ * E_ * 2;      // 2 MB each
  const size_t sz_wo  = (size_t)E_ * E_ * 2;            // 2 MB
  const size_t sz_qkv = (size_t)B_ * H_ * T_ * DH_ * 2; // 8 MB each
  _Float16* xh  = (_Float16*)(ws);
  _Float16* wqt = (_Float16*)(ws + sz_xh);
  _Float16* wkt = (_Float16*)(ws + sz_xh + sz_wt);
  _Float16* wvt = (_Float16*)(ws + sz_xh + 2 * sz_wt);
  _Float16* woh = (_Float16*)(ws + sz_xh + 3 * sz_wt);
  _Float16* Qd  = (_Float16*)(ws + sz_xh + 3 * sz_wt + sz_wo);
  _Float16* Kd  = (_Float16*)(ws + sz_xh + 3 * sz_wt + sz_wo + sz_qkv);
  _Float16* Vtd = (_Float16*)(ws + sz_xh + 3 * sz_wt + sz_wo + 2 * sz_qkv);
  _Float16* Ohd = (_Float16*)(ws + sz_xh + 3 * sz_wt + sz_wo + 3 * sz_qkv);

  const int nx = B_ * T_ * E_;
  const int nw = H_ * E_ * DH_;
  cvt_f32_f16<<<nx / 256, 256, 0, stream>>>(x, xh, nx);
  transpose_w<<<nw / 256, 256, 0, stream>>>(Wq, wqt);
  transpose_w<<<nw / 256, 256, 0, stream>>>(Wk, wkt);
  transpose_w<<<nw / 256, 256, 0, stream>>>(Wv, wvt);
  cvt_f32_f16<<<(E_ * E_) / 256, 256, 0, stream>>>(wo_w, woh, E_ * E_);

  qkv_gemm<<<dim3(T_ / 16, H_, B_), 128, 0, stream>>>(xh, wqt, wkt, wvt, Qd, Kd, Vtd);
  flash_attn<<<dim3(T_ / 64, B_ * H_), 128, 0, stream>>>(Qd, Kd, Vtd, Ohd);
  out_proj<<<dim3((B_ * T_) / 16, E_ / 256), 128, 0, stream>>>(Ohd, woh, wo_b, out);
}
